// LSTM_78812649882388
// MI455X (gfx1250) — matrix operand order repaired and checked by reading, not compiled, not run on hardware
//
#include <hip/hip_runtime.h>

typedef _Float16 v16h __attribute__((ext_vector_type(16)));
typedef float    v8f  __attribute__((ext_vector_type(8)));

// swap lanes l <-> l^16 (full half exchange), group-of-32 swizzle, xor=0x10
#define SWZ16I(val) __builtin_amdgcn_ds_swizzle((val), 0x401f)
#define SWZ16F(val) \
  __builtin_bit_cast(float, __builtin_amdgcn_ds_swizzle(__builtin_bit_cast(int, (val)), 0x401f))

static constexpr int B_TOT = 16384;
static constexpr int T_LEN = 512;
static constexpr int WAVES_PER_BLOCK = 8;
static constexpr int BATCH_PER_WAVE  = 16;

union V16HU { v16h h; int i[8]; };
union H2U   { _Float16 h[2]; int i; };

#if __has_builtin(__builtin_amdgcn_tanhf)
__device__ __forceinline__ float tanh_fast(float x) { return __builtin_amdgcn_tanhf(x); }
#else
__device__ __forceinline__ float tanh_fast(float x) {
  return fmaf(__builtin_amdgcn_rcpf(1.0f + __builtin_amdgcn_exp2f(x * -2.88539008f)), 2.0f, -1.0f);
}
#endif

// Gate-row permutation: A tile rows M=0..15 -> weight rows [i0-3 | f0-3 | i4-7 | f4-7]
// (per tile; tile1 analogous with g/o). perm(n) = n + {0,+4,-4,0}[n>>2].
__device__ __forceinline__ int permRow(int v) {
  const int q = v >> 2;
  return v + ((q == 1) ? 4 : (q == 2) ? -4 : 0);
}

__global__ __launch_bounds__(256)
void lstm_bidir_wmma(const float* __restrict__ x,
                     const float* __restrict__ h0,
                     const float* __restrict__ c0,
                     const float* __restrict__ wih_f,
                     const float* __restrict__ whh_f,
                     const float* __restrict__ bias_f,
                     const float* __restrict__ wih_b,
                     const float* __restrict__ whh_b,
                     const float* __restrict__ bias_b,
                     const float* __restrict__ w_lin,
                     const float* __restrict__ b_lin,
                     float* __restrict__ out)
{
  const int tid    = threadIdx.x;
  const int lane   = tid & 31;
  const int waveId = tid >> 5;
  const int n      = lane & 15;   // batch column (B/C/D); gate row M (A)
  const int hi     = lane >> 4;   // half: owns hidden units 0-3 (lo) / 4-7 (hi)
  const int batch0 = (blockIdx.x * WAVES_PER_BLOCK + waveId) * BATCH_PER_WAVE;
  const int bidx   = batch0 + n;  // this lane's batch row

  // ---- A = permuted extended weight matrix, built once per direction/tile.
  // 16-bit A 16x32 layout: lanes0-15: v0..3 = K0..7 (f16 pairs); lanes16-31: v0 = K8,9.
  // K=0..7 -> Whh row, K=8 -> Wih (input term folded into the GEMM).
  // CRITICAL: all A columns K=9..31 are EXACT ZERO -> B rows K=9..31 are don't-care
  // (any finite value), which lets the per-step B build skip all lane masking.
  auto buildA = [&](const float* whh, const float* wih, int tile) -> V16HU {
    V16HU A;
    const int g = permRow(n) + 16 * tile;
    H2U w01, w23, w45, w67, wx;
    w01.h[0] = (_Float16)whh[g * 8 + 0]; w01.h[1] = (_Float16)whh[g * 8 + 1];
    w23.h[0] = (_Float16)whh[g * 8 + 2]; w23.h[1] = (_Float16)whh[g * 8 + 3];
    w45.h[0] = (_Float16)whh[g * 8 + 4]; w45.h[1] = (_Float16)whh[g * 8 + 5];
    w67.h[0] = (_Float16)whh[g * 8 + 6]; w67.h[1] = (_Float16)whh[g * 8 + 7];
    wx.h[0]  = (_Float16)wih[g];         wx.h[1]  = (_Float16)0.0f;
    A.i[0] = hi ? wx.i : w01.i;   // hi half: K=8,9 = (wih, 0)
    A.i[1] = hi ? 0    : w23.i;   // hi half: K=10,11 = 0
    A.i[2] = hi ? 0    : w45.i;
    A.i[3] = hi ? 0    : w67.i;
    A.i[4] = 0; A.i[5] = 0; A.i[6] = 0; A.i[7] = 0;   // K=16..31 = 0 (both halves)
    return A;
  };

  // Per-VGPR bias registers (C is inline 0 in the WMMAs):
  //   bh0[v]  = 0.5*bias[i/f row]   (tile0, all sigmoid)
  //   bg1[v]  = bias[g row] (v<4, tanh) | 0.5*bias[o row] (v>=4, sigmoid)
  auto buildBias = [&](const float* bias, float* bh0, float* bg1) {
    #pragma unroll
    for (int v = 0; v < 4; ++v) {
      bh0[v]     = 0.5f * bias[hi ? v + 4  : v];            // i_{v+4hi}
      bh0[v + 4] = 0.5f * bias[hi ? v + 12 : v + 8];        // f_{v+4hi}
      bg1[v]     =        bias[16 + (hi ? v + 4  : v)];     // g_{v+4hi}
      bg1[v + 4] = 0.5f * bias[16 + (hi ? v + 12 : v + 8)]; // o_{v+4hi}
    }
  };

  // One LSTM cell step. State is lane-local: cs/hs[u] = c/h[bidx][u + 4*hi].
  auto cell_step = [&](float xv, const V16HU& A0, const V16HU& A1,
                       const float* bh0, const float* bg1,
                       float* cs, float* hs) {
    // B = [h ; x] column: 32x16 f16; lanes0-15 v0..3 = K0..7 (h); lanes16-31 v0 = K8,9 (x,0).
    // hi lanes hold K=8..15,24..31; all but K=8 hit A's zero columns: no masking needed,
    // values are finite (cvt_pkrtz of bounded h and the input x), so no NaN risk.
    const int pk01 = __builtin_bit_cast(int, __builtin_amdgcn_cvt_pkrtz(hs[0], hs[1]));
    const int pk23 = __builtin_bit_cast(int, __builtin_amdgcn_cvt_pkrtz(hs[2], hs[3]));
    const int sw01 = SWZ16I(pk01);  // lo receives hi's (h4,h5) -> K=4,5
    const int sw23 = SWZ16I(pk23);  // lo receives hi's (h6,h7) -> K=6,7
    const int bx   = __builtin_bit_cast(int, __builtin_amdgcn_cvt_pkrtz(xv, 0.0f));
    V16HU Bm;
    Bm.i[0] = hi ? bx : pk01;
    Bm.i[1] = pk23;
    Bm.i[2] = sw01;
    Bm.i[3] = sw23;
    Bm.i[4] = bx;                   // K=16,17 (lo) / K=24,25 (hi) = (x,0), masked by A's zero columns
    Bm.i[5] = 0; Bm.i[6] = 0; Bm.i[7] = 0;

    const v8f zeroC = {};
    // Z^T: D(M=gate-row, N=batch); lane = batch, VGPR v -> permuted gate (v + 8*hi)
    const v8f z0 = __builtin_amdgcn_wmma_f32_16x16x32_f16(false, A0.h, false, Bm.h, (short)0, zeroC, false, false);
    const v8f z1 = __builtin_amdgcn_wmma_f32_16x16x32_f16(false, A1.h, false, Bm.h, (short)0, zeroC, false, false);

    // Fully lane-local gate math: z0[u]=i_u', z0[u+4]=f_u', z1[u]=g_u', z1[u+4]=o_u'
    #pragma unroll
    for (int u = 0; u < 4; ++u) {
      const float si = fmaf(tanh_fast(fmaf(z0[u],     0.5f, bh0[u]    )), 0.5f, 0.5f);
      const float sf = fmaf(tanh_fast(fmaf(z0[u + 4], 0.5f, bh0[u + 4])), 0.5f, 0.5f);
      const float tg =      tanh_fast(     z1[u] +          bg1[u]    );
      const float so = fmaf(tanh_fast(fmaf(z1[u + 4], 0.5f, bg1[u + 4])), 0.5f, 0.5f);
      const float c  = fmaf(sf, cs[u], si * tg);
      cs[u] = c;
      hs[u] = so * tanh_fast(c);
    }
  };

  // ---------- forward direction ----------
  const V16HU Af0 = buildA(whh_f, wih_f, 0);
  const V16HU Af1 = buildA(whh_f, wih_f, 1);
  float bh0f[8], bg1f[8];
  buildBias(bias_f, bh0f, bg1f);

  float hsv[4], csv[4];
  #pragma unroll
  for (int u = 0; u < 4; ++u) {
    hsv[u] = h0[(size_t)bidx * 8 + u + 4 * hi];
    csv[u] = c0[(size_t)bidx * 8 + u + 4 * hi];
  }

  const float* xrow = x + (size_t)bidx * T_LEN;
  float xv = xrow[0];                         // software-pipelined input scalar

  #pragma unroll 1
  for (int t = 0; t < T_LEN - 1; ++t) {
    const float xn = xrow[t + 1];             // unconditional: last iter peeled below
    __builtin_prefetch(xrow + t + 33, 0, 1);  // global_prefetch_b8, speculative
    cell_step(xv, Af0, Af1, bh0f, bg1f, csv, hsv);
    xv = xn;
  }
  cell_step(xv, Af0, Af1, bh0f, bg1f, csv, hsv);

  // ---------- backward direction: single cell step on x[:, T-1] from h0[1], c0[1] ----------
  const V16HU Ab0 = buildA(whh_b, wih_b, 0);
  const V16HU Ab1 = buildA(whh_b, wih_b, 1);
  float bh0b[8], bg1b[8];
  buildBias(bias_b, bh0b, bg1b);

  const float* h0b = h0 + (size_t)B_TOT * 8;
  const float* c0b = c0 + (size_t)B_TOT * 8;
  float hbv[4], cbv[4];
  #pragma unroll
  for (int u = 0; u < 4; ++u) {
    hbv[u] = h0b[(size_t)bidx * 8 + u + 4 * hi];
    cbv[u] = c0b[(size_t)bidx * 8 + u + 4 * hi];
  }
  cell_step(xrow[T_LEN - 1], Ab0, Ab1, bh0b, bg1b, cbv, hbv);

  // ---------- final linear: lane-local partials + one half-exchange ----------
  float p = 0.0f;
  #pragma unroll
  for (int u = 0; u < 4; ++u) {
    p = fmaf(hsv[u], w_lin[u + 4 * hi],     p);
    p = fmaf(hbv[u], w_lin[8 + u + 4 * hi], p);
  }
  p += SWZ16F(p);                 // lo partial (units 0-3) + hi partial (units 4-7)
  if (!hi)
    out[bidx] = p + b_lin[0];
}

extern "C" void kernel_launch(void* const* d_in, const int* in_sizes, int n_in,
                              void* d_out, int out_size, void* d_ws, size_t ws_size,
                              hipStream_t stream) {
  (void)in_sizes; (void)n_in; (void)out_size; (void)d_ws; (void)ws_size;
  const float* x      = (const float*)d_in[0];
  const float* h0     = (const float*)d_in[1];
  const float* c0     = (const float*)d_in[2];
  const float* wih_f  = (const float*)d_in[3];
  const float* whh_f  = (const float*)d_in[4];
  const float* b_f    = (const float*)d_in[5];
  const float* wih_b  = (const float*)d_in[6];
  const float* whh_b  = (const float*)d_in[7];
  const float* b_b    = (const float*)d_in[8];
  const float* w_lin  = (const float*)d_in[9];
  const float* b_lin  = (const float*)d_in[10];
  float* out = (float*)d_out;

  dim3 grid(B_TOT / (BATCH_PER_WAVE * WAVES_PER_BLOCK));  // 128 blocks
  dim3 block(256);                                        // 8 waves of 32 (wave32)
  lstm_bidir_wmma<<<grid, block, 0, stream>>>(x, h0, c0, wih_f, whh_f, b_f,
                                              wih_b, whh_b, b_b, w_lin, b_lin, out);
}
